// DAWN_22282290331741
// MI455X (gfx1250) — matrix operand order repaired and checked by reading, not compiled, not run on hardware
//
#include <hip/hip_runtime.h>
#include <hip/hip_bf16.h>
#include <math.h>

typedef __bf16 bf16;
typedef __attribute__((ext_vector_type(16))) __bf16 v16bf;
typedef __attribute__((ext_vector_type(8)))  __bf16 v8bf;
typedef __attribute__((ext_vector_type(8)))  float   v8f;
typedef __attribute__((ext_vector_type(4)))  float   v4f;

#define D_MODEL 1024
#define DFF_    4096
#define NHEAD   16
#define DHEAD   64
#define NNEUR   512
#define NPAT    256
#define SEQ     1024
#define BATCH   2
#define MTOK    (BATCH*SEQ)
#define TOPK_N  8
#define TOPK_P  16

static __device__ __forceinline__ bf16 f2bf(float x) { return (bf16)x; }

static __device__ __forceinline__ v8f zero_v8f() {
  v8f z = {0.f,0.f,0.f,0.f,0.f,0.f,0.f,0.f};
  return z;
}

static __device__ __forceinline__ v8f wmma_bf16(v16bf a, v16bf b, v8f c) {
  // D = A(16x32 bf16) * B(32x16 bf16) + C(16x16 f32)
  return __builtin_amdgcn_wmma_f32_16x16x32_bf16(false, a, false, b, (short)0, c, false, false);
}

// ---------------------------------------------------------------------------
// GEMM: Y[M,N] = X[M,K] @ W[N,K]^T (+bias) (+=Y if ADD)
// Block tile 128x128, 8 waves (4x2), each wave 32x64 (2x4 WMMA tiles),
// K-step 32, double-buffered LDS (one barrier per K-step).
// f32 global -> bf16 LDS staging; fragments via 16B-aligned ds_load_b128
// matching the documented CDNA5 A/B VGPR layouts.
// Requires M%128==0, N%128==0, K%32==0 (true for all call sites).
// ---------------------------------------------------------------------------
#define SA 40
#define SB 40
template<int ADD, int HASBIAS>
__global__ __launch_bounds__(256) void gemm_bf16_kernel(
    const float* __restrict__ X, const float* __restrict__ W,
    const float* __restrict__ bias, float* __restrict__ Y,
    int M, int N, int K)
{
  __shared__ __align__(16) bf16 As[2][128 * SA];
  __shared__ __align__(16) bf16 Bs[2][128 * SB];

  const int tid  = threadIdx.x;
  const int lane = tid & 31;
  const int wv   = tid >> 5;
  const int lm   = lane & 15;
  const int hf   = lane >> 4;           // 0: lanes 0-15, 1: lanes 16-31
  const int wm   = wv & 3;              // 4 wave rows   -> 32 rows each
  const int wn   = wv >> 2;             // 2 wave cols   -> 64 cols each
  const int m0   = blockIdx.y * 128;
  const int n0   = blockIdx.x * 128;

  v8f acc[2][4];
  #pragma unroll
  for (int i = 0; i < 2; ++i)
    #pragma unroll
    for (int j = 0; j < 4; ++j) acc[i][j] = zero_v8f();

  // staging: both tiles are 128 rows x 32 cols, 16 f32 per thread
  const int srow = tid >> 1, scb = (tid & 1) * 16;
  const int offA = hf ? 8 : 0;
  const int offB = hf ? 8 : 0;

  const float* xsrc = X + (size_t)(m0 + srow) * K + scb;
  const float* wsrc = W + (size_t)(n0 + srow) * K + scb;

  auto stage = [&](int buf, int k0) {
    const v4f* sx = (const v4f*)(xsrc + k0);
    const v4f* sw = (const v4f*)(wsrc + k0);
    bf16* da = &As[buf][srow * SA + scb];
    bf16* db = &Bs[buf][srow * SB + scb];
    #pragma unroll
    for (int c = 0; c < 4; ++c) {
      v4f f = sx[c];
      da[c*4+0] = f2bf(f.x); da[c*4+1] = f2bf(f.y);
      da[c*4+2] = f2bf(f.z); da[c*4+3] = f2bf(f.w);
    }
    #pragma unroll
    for (int c = 0; c < 4; ++c) {
      v4f f = sw[c];
      db[c*4+0] = f2bf(f.x); db[c*4+1] = f2bf(f.y);
      db[c*4+2] = f2bf(f.z); db[c*4+3] = f2bf(f.w);
    }
  };

  stage(0, 0);
  __syncthreads();

  const int nk = K >> 5;
  for (int ks = 0; ks < nk; ++ks) {
    const int cur = ks & 1;
    if (ks + 1 < nk) {
      if (ks + 2 < nk) {   // prefetch two slabs ahead (global_prefetch_b8)
        __builtin_prefetch(xsrc + ((ks + 2) << 5), 0, 1);
        __builtin_prefetch(wsrc + ((ks + 2) << 5), 0, 1);
      }
      stage(cur ^ 1, (ks + 1) << 5);   // fill idle buffer while computing
    }

    v16bf afrag[2], bfrag[4];
    #pragma unroll
    for (int mt = 0; mt < 2; ++mt) {
      const bf16* base = &As[cur][(wm * 32 + mt * 16 + lm) * SA];
      v8bf lo = *(const v8bf*)(base + offA);
      v8bf hi = *(const v8bf*)(base + 16 + offA);
      #pragma unroll
      for (int i = 0; i < 8; ++i) { afrag[mt][i] = lo[i]; afrag[mt][i+8] = hi[i]; }
    }
    #pragma unroll
    for (int nt = 0; nt < 4; ++nt) {
      const bf16* base = &Bs[cur][(wn * 64 + nt * 16 + lm) * SB];
      v8bf lo = *(const v8bf*)(base + offB);
      v8bf hi = *(const v8bf*)(base + 16 + offB);
      #pragma unroll
      for (int i = 0; i < 8; ++i) { bfrag[nt][i] = lo[i]; bfrag[nt][i+8] = hi[i]; }
    }
    #pragma unroll
    for (int mt = 0; mt < 2; ++mt)
      #pragma unroll
      for (int nt = 0; nt < 4; ++nt)
        acc[mt][nt] = wmma_bf16(afrag[mt], bfrag[nt], acc[mt][nt]);
    __syncthreads();
  }

  #pragma unroll
  for (int mt = 0; mt < 2; ++mt) {
    #pragma unroll
    for (int nt = 0; nt < 4; ++nt) {
      const int n = n0 + wn * 64 + nt * 16 + lm;
      const float bv = HASBIAS ? bias[n] : 0.f;
      const int mbase = m0 + wm * 32 + mt * 16 + hf * 8;
      float* yp = Y + (size_t)mbase * N + n;
      #pragma unroll
      for (int r = 0; r < 8; ++r) {
        const float val = acc[mt][nt][r] + bv;
        if (ADD) *yp += val; else *yp = val;
        yp += N;
      }
    }
  }
}

// ---------------------------------------------------------------------------
// Fused causal flash attention. q/k/v/ctx: [B,S,D] with head-major packing
// (d = h*64+dh). One wave handles 16 query rows; 4 waves/block share one (b,h).
// Online softmax; QK^T and PV via WMMA; C->A layout swap through LDS.
// ---------------------------------------------------------------------------
__global__ __launch_bounds__(128) void attn_kernel(
    const float* __restrict__ q, const float* __restrict__ k,
    const float* __restrict__ v, float* __restrict__ ctx)
{
  __shared__ __align__(16) float Sbuf[4][16][36];
  __shared__ __align__(16) bf16  Pbuf[4][16][40];
  __shared__ __align__(16) bf16  Vt[4][64][40];
  __shared__ float rowscale[4][16];
  __shared__ float rowsum_s[4][16];

  const int tid  = threadIdx.x;
  const int lane = tid & 31;
  const int w    = tid >> 5;
  const int lm   = lane & 15;
  const int hf   = lane >> 4;

  const int tile = blockIdx.x;
  const int qb = tile & 15;              // S/64 = 16
  const int h  = (tile >> 4) & 15;       // NH
  const int b  = tile >> 8;              // B
  const int bq0 = qb * 64;
  const int q0  = bq0 + w * 16;
  const float qscale = 0.125f;           // 1/sqrt(64)

  // Q fragments (A-layout, pre-scaled)
  v16bf aQ[2];
  {
    const float* qrow = q + (size_t)(b * SEQ + q0 + lm) * D_MODEL + h * DHEAD;
    const int off = hf ? 8 : 0;
    #pragma unroll
    for (int kk = 0; kk < 2; ++kk)
      #pragma unroll
      for (int i = 0; i < 8; ++i) {
        aQ[kk][i]   = f2bf(qrow[kk*32 + off + i] * qscale);
        aQ[kk][i+8] = f2bf(qrow[kk*32 + 16 + off + i] * qscale);
      }
  }

  v8f o[4];
  #pragma unroll
  for (int ot = 0; ot < 4; ++ot) o[ot] = zero_v8f();
  float rmax = -3.0e38f, rsum = 0.f;

  const int nkb = (bq0 + 64) >> 5;       // uniform across block -> barriers legal
  for (int kb = 0; kb < nkb; ++kb) {
    const int kb32 = kb << 5;

    // stage V tile transposed: Vt[dh][j] = V[kb32+j][dh]
    {
      const float* vrow = v + (size_t)(b * SEQ + kb32 + lane) * D_MODEL + h * DHEAD;
      for (int dh = 0; dh < 64; ++dh) Vt[w][dh][lane] = f2bf(vrow[dh]);
    }

    // S = Q @ K^T for 32 keys (two 16-key column tiles)
    v8f st[2]; st[0] = zero_v8f(); st[1] = zero_v8f();
    #pragma unroll
    for (int nt = 0; nt < 2; ++nt) {
      const float* krow = k + (size_t)(b * SEQ + kb32 + nt*16 + lm) * D_MODEL + h * DHEAD;
      const int boff = hf ? 8 : 0;
      #pragma unroll
      for (int kk = 0; kk < 2; ++kk) {
        v16bf bK;
        #pragma unroll
        for (int i = 0; i < 16; ++i) bK[i] = f2bf(krow[kk*32 + boff + (i & 7) + (i >> 3) * 16]);
        st[nt] = wmma_bf16(aQ[kk], bK, st[nt]);
      }
    }

    // causal mask + spill to LDS (C-layout)
    #pragma unroll
    for (int nt = 0; nt < 2; ++nt) {
      const int key = kb32 + nt * 16 + lm;
      #pragma unroll
      for (int r = 0; r < 8; ++r) {
        const int qi = q0 + r + hf * 8;
        Sbuf[w][r + hf*8][nt*16 + lm] = (key <= qi) ? st[nt][r] : -3.0e38f;
      }
    }
    __syncthreads();

    // online softmax stats per row (lane handles row lm; halves duplicate)
    {
      float mnew = rmax;
      for (int j = 0; j < 32; ++j) mnew = fmaxf(mnew, Sbuf[w][lm][j]);
      const float scl = __expf(rmax - mnew);
      float ps = 0.f;
      for (int j = 0; j < 32; ++j) {
        const float p = __expf(Sbuf[w][lm][j] - mnew);
        ps += p;
        if (hf == 0) Pbuf[w][lm][j] = f2bf(p);
      }
      rsum = rsum * scl + ps;
      rmax = mnew;
      if (hf == 0) rowscale[w][lm] = scl;
    }
    __syncthreads();

    // rescale accumulators
    #pragma unroll
    for (int ot = 0; ot < 4; ++ot)
      #pragma unroll
      for (int r = 0; r < 8; ++r) o[ot][r] *= rowscale[w][r + hf*8];

    // P fragment (A-layout) from LDS
    v16bf pf;
    {
      const int off = hf ? 8 : 0;
      const bf16* base = &Pbuf[w][lm][0];
      v8bf lo = *(const v8bf*)(base + off);
      v8bf hi = *(const v8bf*)(base + 16 + off);
      #pragma unroll
      for (int i = 0; i < 8; ++i) { pf[i] = lo[i]; pf[i+8] = hi[i]; }
    }
    // O += P @ V  (B-frags contiguous from transposed V tile)
    {
      const int boff = hf ? 8 : 0;
      #pragma unroll
      for (int ot = 0; ot < 4; ++ot) {
        const bf16* base = &Vt[w][ot*16 + lm][0];
        v8bf lo = *(const v8bf*)(base + boff);
        v8bf hi = *(const v8bf*)(base + 16 + boff);
        v16bf bV;
        #pragma unroll
        for (int i = 0; i < 8; ++i) { bV[i] = lo[i]; bV[i+8] = hi[i]; }
        o[ot] = wmma_bf16(pf, bV, o[ot]);
      }
    }
    __syncthreads();
  }

  if (hf == 0) rowsum_s[w][lm] = rsum;
  __syncthreads();

  #pragma unroll
  for (int ot = 0; ot < 4; ++ot)
    #pragma unroll
    for (int r = 0; r < 8; ++r) {
      const float inv = 1.f / rowsum_s[w][r + hf*8];
      const int m = q0 + r + hf * 8;
      ctx[(size_t)(b * SEQ + m) * D_MODEL + h * DHEAD + ot*16 + lm] = o[ot][r] * inv;
    }
}

// ---------------------------------------------------------------------------
// Small per-token kernels
// ---------------------------------------------------------------------------
__global__ __launch_bounds__(256) void embed_kernel(
    const int* __restrict__ ids, const float* __restrict__ tok,
    const float* __restrict__ pos, float* __restrict__ x)
{
  const int i = blockIdx.x * 256 + threadIdx.x;      // over M*D (exact)
  const int m = i >> 10, dd = i & 1023;
  const int s = m & (SEQ - 1);
  x[i] = tok[(size_t)ids[m] * D_MODEL + dd] + pos[(size_t)s * D_MODEL + dd];
}

__global__ __launch_bounds__(256) void ln_kernel(
    const float* __restrict__ x, const float* __restrict__ g,
    const float* __restrict__ bta, float* __restrict__ out)
{
  __shared__ float red[256];
  const int m = blockIdx.x, tid = threadIdx.x;
  const float* row = x + (size_t)m * D_MODEL;
  float s = 0.f;
  for (int i = tid; i < D_MODEL; i += 256) s += row[i];
  red[tid] = s; __syncthreads();
  for (int st = 128; st > 0; st >>= 1) { if (tid < st) red[tid] += red[tid + st]; __syncthreads(); }
  const float mean = red[0] * (1.f / D_MODEL);
  __syncthreads();
  float vs = 0.f;
  for (int i = tid; i < D_MODEL; i += 256) { float d = row[i] - mean; vs += d * d; }
  red[tid] = vs; __syncthreads();
  for (int st = 128; st > 0; st >>= 1) { if (tid < st) red[tid] += red[tid + st]; __syncthreads(); }
  const float inv = rsqrtf(red[0] * (1.f / D_MODEL) + 1e-5f);
  for (int i = tid; i < D_MODEL; i += 256)
    out[(size_t)m * D_MODEL + i] = (row[i] - mean) * inv * g[i] + bta[i];
}

// 2-way path softmax over Linear(concat[a,c]) then comb = w0*s0 + w1*s1
__global__ __launch_bounds__(256) void path_combine_kernel(
    const float* __restrict__ a, const float* __restrict__ c,
    const float* __restrict__ W, const float* __restrict__ bias,
    const float* __restrict__ s0, const float* __restrict__ s1,
    float* __restrict__ comb, int NS)
{
  __shared__ float r0[256], r1[256];
  __shared__ float wsh[2];
  const int m = blockIdx.x, tid = threadIdx.x;
  const float* av = a + (size_t)m * D_MODEL;
  const float* cv = c + (size_t)m * D_MODEL;
  float p0 = 0.f, p1 = 0.f;
  for (int i = tid; i < D_MODEL; i += 256) {
    const float x0 = av[i], x1 = cv[i];
    p0 += x0 * W[i]              + x1 * W[D_MODEL + i];
    p1 += x0 * W[2*D_MODEL + i]  + x1 * W[3*D_MODEL + i];
  }
  r0[tid] = p0; r1[tid] = p1; __syncthreads();
  for (int st = 128; st > 0; st >>= 1) {
    if (tid < st) { r0[tid] += r0[tid + st]; r1[tid] += r1[tid + st]; }
    __syncthreads();
  }
  if (tid == 0) {
    const float l0 = r0[0] + bias[0], l1 = r1[0] + bias[1];
    const float mx = fmaxf(l0, l1);
    const float e0 = __expf(l0 - mx), e1 = __expf(l1 - mx);
    const float inv = 1.f / (e0 + e1);
    wsh[0] = e0 * inv; wsh[1] = e1 * inv;
  }
  __syncthreads();
  const float w0 = wsh[0], w1 = wsh[1];
  for (int j = tid; j < NS; j += 256)
    comb[(size_t)m * NS + j] = w0 * s0[(size_t)m * NS + j] + w1 * s1[(size_t)m * NS + j];
}

__global__ __launch_bounds__(256) void router_topk_kernel(
    const float* __restrict__ comb, const float* __restrict__ neurons,
    float* __restrict__ router_out, float* __restrict__ x, float* __restrict__ sel)
{
  __shared__ float sc[NNEUR];
  __shared__ float rv[256]; __shared__ int ri[256];
  __shared__ float kw[TOPK_N]; __shared__ int ki[TOPK_N];
  const int m = blockIdx.x, tid = threadIdx.x;
  for (int j = tid; j < NNEUR; j += 256) sc[j] = comb[(size_t)m * NNEUR + j];
  __syncthreads();
  for (int t = 0; t < TOPK_N; ++t) {
    float bv = -3.0e38f; int bi = 0;
    for (int j = tid; j < NNEUR; j += 256) { const float v0 = sc[j]; if (v0 > bv) { bv = v0; bi = j; } }
    rv[tid] = bv; ri[tid] = bi; __syncthreads();
    for (int st = 128; st > 0; st >>= 1) {
      if (tid < st && rv[tid + st] > rv[tid]) { rv[tid] = rv[tid + st]; ri[tid] = ri[tid + st]; }
      __syncthreads();
    }
    if (tid == 0) { ki[t] = ri[0]; kw[t] = rv[0]; sc[ri[0]] = -3.0e38f; }
    __syncthreads();
  }
  if (tid == 0) {
    const float mx = kw[0];
    float ssum = 0.f; float e[TOPK_N];
    for (int t = 0; t < TOPK_N; ++t) { e[t] = __expf(kw[t] - mx); ssum += e[t]; }
    for (int t = 0; t < TOPK_N; ++t) kw[t] = e[t] / ssum;
  }
  __syncthreads();
  for (int dd = tid; dd < D_MODEL; dd += 256) {
    float acc = 0.f;
    #pragma unroll
    for (int t = 0; t < TOPK_N; ++t) acc += kw[t] * neurons[(size_t)ki[t] * D_MODEL + dd];
    router_out[(size_t)m * D_MODEL + dd] = acc;
    x[(size_t)m * D_MODEL + dd] += acc;
  }
  for (int j = tid; j < NNEUR; j += 256) sel[(size_t)m * NNEUR + j] = 0.f;
  __syncthreads();
  if (tid < TOPK_N) sel[(size_t)m * NNEUR + ki[tid]] = kw[tid];
}

__global__ __launch_bounds__(256) void ffn_topk_gate_kernel(
    const float* __restrict__ comb2, const float* __restrict__ gates,
    float* __restrict__ gate_out)
{
  __shared__ float sc[NPAT];
  __shared__ float rv[256]; __shared__ int ri[256];
  __shared__ float kw[TOPK_P]; __shared__ int ki[TOPK_P];
  const int m = blockIdx.x, tid = threadIdx.x;
  for (int j = tid; j < NPAT; j += 256) sc[j] = comb2[(size_t)m * NPAT + j];
  __syncthreads();
  for (int t = 0; t < TOPK_P; ++t) {
    float bv = -3.0e38f; int bi = 0;
    if (tid < NPAT) { bv = sc[tid]; bi = tid; }
    rv[tid] = bv; ri[tid] = bi; __syncthreads();
    for (int st = 128; st > 0; st >>= 1) {
      if (tid < st && rv[tid + st] > rv[tid]) { rv[tid] = rv[tid + st]; ri[tid] = ri[tid + st]; }
      __syncthreads();
    }
    if (tid == 0) { ki[t] = ri[0]; kw[t] = rv[0]; sc[ri[0]] = -3.0e38f; }
    __syncthreads();
  }
  if (tid == 0) {
    const float mx = kw[0];
    float ssum = 0.f; float e[TOPK_P];
    for (int t = 0; t < TOPK_P; ++t) { e[t] = __expf(kw[t] - mx); ssum += e[t]; }
    for (int t = 0; t < TOPK_P; ++t) kw[t] = e[t] / ssum;
  }
  __syncthreads();
  for (int f = tid; f < DFF_; f += 256) {
    float acc = 0.f;
    #pragma unroll
    for (int t = 0; t < TOPK_P; ++t) acc += kw[t] * gates[(size_t)ki[t] * DFF_ + f];
    gate_out[(size_t)m * DFF_ + f] = acc;
  }
}

__global__ __launch_bounds__(256) void act_kernel(
    float* __restrict__ hbuf, const float* __restrict__ gate, int n)
{
  const int i = blockIdx.x * 256 + threadIdx.x;
  if (i < n) {
    const float g = gate[i];
    const float sg = 1.f / (1.f + __expf(-g));
    const float z = hbuf[i] * sg;
    hbuf[i] = 0.5f * z * (1.f + erff(z * 0.70710678f));
  }
}

// ---------------------------------------------------------------------------
// Host orchestration
// ---------------------------------------------------------------------------
enum { NEURONS=0, QW, QB, KW, KB, VW, VB, RPW, RPB, N1G, N1B, N2G, N2B,
       PATTERNS, GATES, FPW, FPB, UPW, UPB, DOWNW, DOWNB, CONNW };

static void launch_gemm(const float* X, const float* Wt, const float* bias,
                        float* Y, int Mv, int Nv, int Kv, int add,
                        hipStream_t stream) {
  dim3 g(Nv / 128, Mv / 128);
  if (add) {
    if (bias) gemm_bf16_kernel<1,1><<<g, 256, 0, stream>>>(X, Wt, bias, Y, Mv, Nv, Kv);
    else      gemm_bf16_kernel<1,0><<<g, 256, 0, stream>>>(X, Wt, bias, Y, Mv, Nv, Kv);
  } else {
    if (bias) gemm_bf16_kernel<0,1><<<g, 256, 0, stream>>>(X, Wt, bias, Y, Mv, Nv, Kv);
    else      gemm_bf16_kernel<0,0><<<g, 256, 0, stream>>>(X, Wt, bias, Y, Mv, Nv, Kv);
  }
}

extern "C" void kernel_launch(void* const* d_in, const int* in_sizes, int n_in,
                              void* d_out, int out_size, void* d_ws, size_t ws_size,
                              hipStream_t stream) {
  (void)in_sizes; (void)n_in; (void)out_size; (void)ws_size;
  const int M = MTOK;

  const int*   ids     = (const int*)d_in[0];
  const float* tok_emb = (const float*)d_in[1];
  const float* pos_emb = (const float*)d_in[2];
  const float* norm_g  = (const float*)d_in[3];
  const float* norm_b  = (const float*)d_in[4];

  const float* lp[4][22];
  int p = 5;
  for (int i = 0; i < 4; ++i) {
    for (int j = 0; j < 21; ++j) lp[i][j] = (const float*)d_in[p++];
    lp[i][CONNW] = (i > 0) ? (const float*)d_in[p++] : nullptr;
  }

  float* w = (float*)d_ws;
  float* x          = w; w += (size_t)M * D_MODEL;
  float* nm         = w; w += (size_t)M * D_MODEL;
  float* qb_        = w; w += (size_t)M * D_MODEL;
  float* kb_        = w; w += (size_t)M * D_MODEL;
  float* vb_        = w; w += (size_t)M * D_MODEL;
  float* ctx        = w; w += (size_t)M * D_MODEL;
  float* rout       = w; w += (size_t)M * D_MODEL;
  float* tok_s      = w; w += (size_t)M * NNEUR;
  float* ctx_s      = w; w += (size_t)M * NNEUR;
  float* comb       = w; w += (size_t)M * NNEUR;
  float* selA       = w; w += (size_t)M * NNEUR;
  float* selB       = w; w += (size_t)M * NNEUR;
  float* p_s        = w; w += (size_t)M * NPAT;
  float* r_s        = w; w += (size_t)M * NPAT;
  float* comb2      = w; w += (size_t)M * NPAT;
  float* gate       = w; w += (size_t)M * DFF_;
  float* hb         = w; w += (size_t)M * DFF_;

  embed_kernel<<<(M * D_MODEL) / 256, 256, 0, stream>>>(ids, tok_emb, pos_emb, x);

  float* sel_cur = selA;
  float* sel_prev = selB;

  for (int i = 0; i < 4; ++i) {
    const float** L = lp[i];
    // attention block
    ln_kernel<<<M, 256, 0, stream>>>(x, L[N1G], L[N1B], nm);
    launch_gemm(nm, L[QW], L[QB], qb_, M, D_MODEL, D_MODEL, 0, stream);
    launch_gemm(nm, L[KW], L[KB], kb_, M, D_MODEL, D_MODEL, 0, stream);
    launch_gemm(nm, L[VW], L[VB], vb_, M, D_MODEL, D_MODEL, 0, stream);
    attn_kernel<<<BATCH * NHEAD * (SEQ / 64), 128, 0, stream>>>(qb_, kb_, vb_, ctx);
    // neuron routing
    launch_gemm(nm,  L[NEURONS], nullptr, tok_s, M, NNEUR, D_MODEL, 0, stream);
    launch_gemm(ctx, L[NEURONS], nullptr, ctx_s, M, NNEUR, D_MODEL, 0, stream);
    path_combine_kernel<<<M, 256, 0, stream>>>(nm, ctx, L[RPW], L[RPB],
                                               tok_s, ctx_s, comb, NNEUR);
    if (i > 0) launch_gemm(sel_prev, L[CONNW], nullptr, comb, M, NNEUR, NNEUR, 1, stream);
    router_topk_kernel<<<M, 256, 0, stream>>>(comb, L[NEURONS], rout, x, sel_cur);
    // pattern FFN
    ln_kernel<<<M, 256, 0, stream>>>(x, L[N2G], L[N2B], nm);
    launch_gemm(nm,   L[PATTERNS], nullptr, p_s, M, NPAT, D_MODEL, 0, stream);
    launch_gemm(rout, L[PATTERNS], nullptr, r_s, M, NPAT, D_MODEL, 0, stream);
    path_combine_kernel<<<M, 256, 0, stream>>>(nm, rout, L[FPW], L[FPB],
                                               p_s, r_s, comb2, NPAT);
    ffn_topk_gate_kernel<<<M, 256, 0, stream>>>(comb2, L[GATES], gate);
    launch_gemm(nm, L[UPW], L[UPB], hb, M, DFF_, D_MODEL, 0, stream);
    act_kernel<<<(M * DFF_) / 256, 256, 0, stream>>>(hb, gate, M * DFF_);
    launch_gemm(hb, L[DOWNW], L[DOWNB], x, M, D_MODEL, DFF_, 1, stream);
    // swap selection ping-pong
    float* t = sel_prev; sel_prev = sel_cur; sel_cur = t;
  }

  // final LN + tied head
  ln_kernel<<<M, 256, 0, stream>>>(x, norm_g, norm_b, nm);
  launch_gemm(nm, tok_emb, nullptr, (float*)d_out, M, 32000, D_MODEL, 0, stream);
}
